// FinalLayersGRU_28097676051055
// MI455X (gfx1250) — compile-verified, matrix operand order checked by reading, not run on hardware
//
#include <hip/hip_runtime.h>
#include <hip/hip_bf16.h>
#include <math.h>

#define B_ 128
#define T_ 512
#define D_ 256
#define H_ 512
#define G_ (3 * H_)
#define KS_ (H_ / 32)   // 16 k-slabs in the recurrent GEMM
#define NWG_SCAN 256    // persistent scan: 8 row-tiles x 32 col-tiles, 1 wave each

typedef __attribute__((ext_vector_type(16))) _Float16 v16h;
typedef __attribute__((ext_vector_type(8)))  _Float16 v8h;
typedef __attribute__((ext_vector_type(8)))  float    v8f;
typedef __attribute__((ext_vector_type(4)))  unsigned int v4u;
typedef __attribute__((ext_vector_type(8)))  int      v8i;
typedef __attribute__((ext_vector_type(4)))  int      v4i;

// ---------------------------------------------------------------------------
// A-fragment gather from row-major f32 (cold path: MLP GEMM only).
// a[i]: k = (i/8)*16 + ((lane>>4)<<3) + ((i>>1)&3)*2 + (i&1), m = lane&15.
// ---------------------------------------------------------------------------
__device__ __forceinline__ v16h load_a_frag_f32(const float* A, int ld, int lane) {
  v16h a;
  int m = lane & 15;
  int khalf = (lane >> 4) << 3;
  const float* row = A + (size_t)m * ld;
#pragma unroll
  for (int g = 0; g < 2; ++g) {
#pragma unroll
    for (int s = 0; s < 4; ++s) {
      int k = g * 16 + khalf + s * 2;
      float2 f = *(const float2*)(row + k);
      a[g * 8 + s * 2]     = (_Float16)f.x;
      a[g * 8 + s * 2 + 1] = (_Float16)f.y;
    }
  }
  return a;
}

// B-fragment gather from row-major f32 (K x N): B layout = A layout of B^T.
__device__ __forceinline__ v16h load_b_frag_f32(const float* Bm, int ld, int lane) {
  v16h b;
  int n = lane & 15;
  int khalf = (lane >> 4) << 3;
#pragma unroll
  for (int g = 0; g < 2; ++g) {
#pragma unroll
    for (int s = 0; s < 4; ++s) {
      int k = g * 16 + khalf + s * 2;
      b[g * 8 + s * 2]     = (_Float16)Bm[(size_t)(k)     * ld + n];
      b[g * 8 + s * 2 + 1] = (_Float16)Bm[(size_t)(k + 1) * ld + n];
    }
  }
  return b;
}

// ---------------------------------------------------------------------------
// Pack row-major f32 (K x N) weights into fragment-native f16 B blocks:
// dst[((nt*(K/32)+ks)*32 + lane)*16 + i]
// ---------------------------------------------------------------------------
__global__ void pack_b_kernel(const float* __restrict__ src,
                              _Float16* __restrict__ dst, int K, int N) {
  int tid = blockIdx.x * blockDim.x + threadIdx.x;
  if (tid >= K * N) return;
  int i    = tid & 15;
  int lane = (tid >> 4) & 31;
  int tile = tid >> 9;
  int ksteps = K >> 5;
  int ks = tile % ksteps;
  int nt = tile / ksteps;
  int n = nt * 16 + (lane & 15);
  int g = i >> 3, s = (i >> 1) & 3, half = i & 1;
  int khalf = (lane >> 4) << 3;
  int k = ks * 32 + g * 16 + khalf + s * 2 + half;
  dst[tid] = (_Float16)src[(size_t)k * N + n];
}

// ---------------------------------------------------------------------------
// Pack x (B,T,D) into fragment-native f16 A blocks for phase 1:
// dst[(((t*8+mt)*(D/32)+ks)*32 + lane)*16 + i], rows = 16 batches at fixed t.
// ---------------------------------------------------------------------------
__global__ void pack_a_x_kernel(const float* __restrict__ x,
                                _Float16* __restrict__ dst) {
  int tid = blockIdx.x * blockDim.x + threadIdx.x;   // over T*B*D
  int i    = tid & 15;
  int lane = (tid >> 4) & 31;
  int tile = tid >> 9;          // rt*(D/32) + ks
  int ks = tile & 7;            // D/32 = 8
  int rt = tile >> 3;
  int t  = rt >> 3;
  int mt = rt & 7;
  int m = lane & 15;
  int khalf = (lane >> 4) << 3;
  int g = i >> 3, s = (i >> 1) & 3, half = i & 1;
  int k = ks * 32 + g * 16 + khalf + s * 2 + half;
  int b = mt * 16 + m;
  dst[tid] = (_Float16)x[((size_t)b * T_ + t) * D_ + k];
}

__global__ void zero_u32(unsigned int* __restrict__ p, int n) {
  int i = blockIdx.x * blockDim.x + threadIdx.x;
  if (i < n) p[i] = 0u;
}

// ---------------------------------------------------------------------------
// Phase 1: xg = x @ gru_kernel + bias0, 2x2 tiles per wave, all operands
// pre-packed f16 -> inner loop is pure b128 loads + WMMA.
// Output in scan-consumption layout:
//   xgp[((((t*8+mt)*3 + gate)*32 + jt)*32 + lane)*8 + i]
// grid = (T*(B/16)/2, G/(16*8)) = (2048, 12), block = 128 (4 waves).
// ---------------------------------------------------------------------------
__global__ void __launch_bounds__(128)
xg_gemm_kernel(const _Float16* __restrict__ xp, const _Float16* __restrict__ kp,
               const float* __restrict__ bias0, _Float16* __restrict__ xgp) {
  int lane = threadIdx.x & 31;
  int wave = threadIdx.x >> 5;
  int rt0 = blockIdx.x * 2;            // row tiles rt0, rt0+1 share t
  int t   = rt0 >> 3;
  int mt0 = rt0 & 7;
  int nt0 = (blockIdx.y * 4 + wave) * 2;   // col tiles nt0, nt0+1
  v8f a00 = {}, a01 = {}, a10 = {}, a11 = {};
  const _Float16* A0 = xp + (size_t)rt0 * 8 * 512;
  const _Float16* A1 = A0 + 8 * 512;
  const _Float16* B0 = kp + (size_t)nt0 * 8 * 512;
  const _Float16* B1 = B0 + 8 * 512;
#pragma unroll
  for (int ks = 0; ks < D_ / 32; ++ks) {
    v16h fa0 = *(const v16h*)(A0 + ((size_t)ks * 32 + lane) * 16);
    v16h fa1 = *(const v16h*)(A1 + ((size_t)ks * 32 + lane) * 16);
    v16h fb0 = *(const v16h*)(B0 + ((size_t)ks * 32 + lane) * 16);
    v16h fb1 = *(const v16h*)(B1 + ((size_t)ks * 32 + lane) * 16);
    a00 = __builtin_amdgcn_wmma_f32_16x16x32_f16(false, fa0, false, fb0, (short)0, a00, false, false);
    a01 = __builtin_amdgcn_wmma_f32_16x16x32_f16(false, fa0, false, fb1, (short)0, a01, false, false);
    a10 = __builtin_amdgcn_wmma_f32_16x16x32_f16(false, fa1, false, fb0, (short)0, a10, false, false);
    a11 = __builtin_amdgcn_wmma_f32_16x16x32_f16(false, fa1, false, fb1, (short)0, a11, false, false);
  }
  int n = lane & 15;
  auto store_tile = [&](int mt, int nt, const v8f& acc) {
    int gate = nt >> 5, jt = nt & 31;
    float bz = bias0[nt * 16 + n];
    v8h pk;
#pragma unroll
    for (int i = 0; i < 8; ++i) pk[i] = (_Float16)(acc[i] + bz);
    size_t o = (((((size_t)t * 8 + mt) * 3 + gate) * 32 + jt) * 32 + lane) * 8;
    *(v8h*)(xgp + o) = pk;
  };
  store_tile(mt0,     nt0,     a00);
  store_tile(mt0,     nt0 + 1, a01);
  store_tile(mt0 + 1, nt0,     a10);
  store_tile(mt0 + 1, nt0 + 1, a11);
}

// ---------------------------------------------------------------------------
// Persistent GRU scan. 256 workgroups x 1 wave; wave (mt,jt) owns the same
// 16x16 h-tile for all 512 steps:
//  - rec_kernel B fragments (48 KB) DMA'd into LDS once via the Tensor Data
//    Mover (TENSOR_LOAD_TO_LDS + s_wait_tensorcnt), fallback: manual staging
//  - f32 h state lives in registers (hreg[8]); only the f16 fragment-packed
//    view of h ping-pongs through global memory for the GEMM
//  - device-scope atomic counter barrier between steps
// ---------------------------------------------------------------------------
__global__ void __launch_bounds__(32)
gru_scan_kernel(_Float16* __restrict__ hfA, _Float16* __restrict__ hfB,
                const _Float16* __restrict__ recKp,
                const float* __restrict__ bias1,
                const _Float16* __restrict__ xgp,
                const int* __restrict__ mask,
                float* __restrict__ hfinal,
                unsigned int* __restrict__ bar) {
  int lane = threadIdx.x;
  int wgid = blockIdx.x;
  int mt = wgid >> 5;                  // 0..7  row tile
  int jt = wgid & 31;                  // 0..31 col tile

  __shared__ _Float16 ldsB[3 * KS_ * 32 * 16];   // 48 KB

#if __has_builtin(__builtin_amdgcn_tensor_load_to_lds)
  // --- TDM staging: one 2D tensor load, tile = 8192 x 3 elements (2B each),
  //     dim0-stride = 262144 elements (gap between the 3 gate chunks). ---
  {
    unsigned long long ga =
        (unsigned long long)(const void*)(recKp + (size_t)jt * (KS_ * 32 * 16));
    unsigned int ldsoff = (unsigned int)(unsigned long long)(const void*)&ldsB[0];
    v4u g0;
    g0[0] = 1u;                                  // count=1, user descriptor
    g0[1] = ldsoff;                              // lds_addr (bytes)
    g0[2] = (unsigned int)ga;                    // global_addr[31:0]
    g0[3] = ((unsigned int)(ga >> 32) & 0x01FFFFFFu) | 0x80000000u; // ga[56:32] | type=2
    v8i g1;
    g1[0] = 0x10000;                             // wg_mask=0, data_size=1 (2 bytes)
    g1[1] = 0;                                   // tensor_dim0[15:0]=0 (262144)
    g1[2] = 0x00030004;                          // tensor_dim0[31:16]=4, tensor_dim1=3
    g1[3] = (int)0x20000000;                     // tile_dim0 = 8192
    g1[4] = 3;                                   // tile_dim1 = 3
    g1[5] = 0x40000;                             // tensor_dim0_stride = 262144
    g1[6] = 0;
    g1[7] = 0;
    v4i gz = {0, 0, 0, 0};
#if defined(__clang_major__) && (__clang_major__ >= 23)
    v8i gz8 = {0, 0, 0, 0, 0, 0, 0, 0};
    __builtin_amdgcn_tensor_load_to_lds(g0, g1, gz, gz, gz8, 0);
#else
    __builtin_amdgcn_tensor_load_to_lds(g0, g1, gz, gz, 0);
#endif
    __builtin_amdgcn_s_wait_tensorcnt(0);
    asm volatile("" ::: "memory");               // LDS now holds the weights
  }
#else
  // --- Fallback: manual staging of 48 B fragments. ---
#pragma unroll
  for (int g = 0; g < 3; ++g)
#pragma unroll 4
    for (int ks = 0; ks < KS_; ++ks) {
      v16h b = *(const v16h*)(recKp +
                 (((size_t)(g * 32 + jt) * KS_ + ks) * 32 + lane) * 16);
      *(v16h*)&ldsB[((g * KS_ + ks) * 32 + lane) * 16] = b;
    }
#endif

  int n = lane & 15;
  int hi = lane >> 4;
  int j = jt * 16 + n;
  float b1z = bias1[j], b1r = bias1[H_ + j], b1h = bias1[2 * H_ + j];

  float hreg[8];
#pragma unroll
  for (int i = 0; i < 8; ++i) hreg[i] = 0.f;

  int g2  = jt & 1;                    // packed-A slot constants (k = j)
  int ks2 = jt >> 1;

  for (int t = 0; t < T_; ++t) {
    const _Float16* hfc = (t & 1) ? hfB : hfA;
    _Float16*       hfn = (t & 1) ? hfA : hfB;

    v8f az = {}, ar = {}, ah = {};
#pragma unroll 4
    for (int ks = 0; ks < KS_; ++ks) {
      v16h a  = *(const v16h*)(hfc + (((size_t)mt * KS_ + ks) * 32 + lane) * 16);
      v16h bz = *(const v16h*)&ldsB[((0 * KS_ + ks) * 32 + lane) * 16];
      v16h br = *(const v16h*)&ldsB[((1 * KS_ + ks) * 32 + lane) * 16];
      v16h bh = *(const v16h*)&ldsB[((2 * KS_ + ks) * 32 + lane) * 16];
      az = __builtin_amdgcn_wmma_f32_16x16x32_f16(false, a, false, bz, (short)0, az, false, false);
      ar = __builtin_amdgcn_wmma_f32_16x16x32_f16(false, a, false, br, (short)0, ar, false, false);
      ah = __builtin_amdgcn_wmma_f32_16x16x32_f16(false, a, false, bh, (short)0, ah, false, false);
    }

    size_t basetm = ((size_t)t * 8 + mt) * 3;
    v8h xz8 = *(const v8h*)(xgp + ((basetm + 0) * 32 + jt) * 256 + (size_t)lane * 8);
    v8h xr8 = *(const v8h*)(xgp + ((basetm + 1) * 32 + jt) * 256 + (size_t)lane * 8);
    v8h xh8 = *(const v8h*)(xgp + ((basetm + 2) * 32 + jt) * 256 + (size_t)lane * 8);

#pragma unroll
    for (int i = 0; i < 8; ++i) {
      int m = i + (hi << 3);
      int brow = mt * 16 + m;
      float z  = 1.f / (1.f + __expf(-((float)xz8[i] + az[i] + b1z)));
      float r  = 1.f / (1.f + __expf(-((float)xr8[i] + ar[i] + b1r)));
      float hh = tanhf((float)xh8[i] + r * (ah[i] + b1h));
      float hn = z * hreg[i] + (1.f - z) * hh;
      hreg[i] = (mask[brow * T_ + t] != 0) ? hn : hreg[i];
      int lane2 = m + (((n >> 3) & 1) << 4);
      int i2 = g2 * 8 + ((n & 7) >> 1) * 2 + (n & 1);
      hfn[(((size_t)mt * KS_ + ks2) * 32 + lane2) * 16 + i2] = (_Float16)hreg[i];
      if (t == T_ - 1) hfinal[(size_t)brow * H_ + j] = hreg[i];
    }

    if (t != T_ - 1) {
      __threadfence();
      if (threadIdx.x == 0) {          // divergent spin holds the whole wave
        __hip_atomic_fetch_add(bar, 1u, __ATOMIC_RELEASE, __HIP_MEMORY_SCOPE_AGENT);
        unsigned tgt = (unsigned)NWG_SCAN * (unsigned)(t + 1);
        while (__hip_atomic_load(bar, __ATOMIC_ACQUIRE, __HIP_MEMORY_SCOPE_AGENT) < tgt)
          __builtin_amdgcn_s_sleep(1);
      }
      __builtin_amdgcn_wave_barrier();
    }
  }
}

// ---------------------------------------------------------------------------
// MLP head part 1: h1 = gelu(h @ w1 + b1), exact erf GELU.
// ---------------------------------------------------------------------------
__global__ void __launch_bounds__(128)
mlp1_kernel(const float* __restrict__ h, const float* __restrict__ w1,
            const float* __restrict__ b1, float* __restrict__ h1) {
  int lane = threadIdx.x & 31;
  int wave = threadIdx.x >> 5;
  int m0 = blockIdx.x * 16;
  int nt = blockIdx.y * 4 + wave;
  v8f acc = {};
#pragma unroll 4
  for (int ks = 0; ks < H_ / 32; ++ks) {
    v16h a = load_a_frag_f32(h + (size_t)m0 * H_ + ks * 32, H_, lane);
    v16h b = load_b_frag_f32(w1 + (size_t)ks * 32 * H_ + nt * 16, H_, lane);
    acc = __builtin_amdgcn_wmma_f32_16x16x32_f16(false, a, false, b,
                                                 (short)0, acc, false, false);
  }
  int n = lane & 15;
  float bb = b1[nt * 16 + n];
#pragma unroll
  for (int i = 0; i < 8; ++i) {
    int m = i + ((lane >> 4) << 3);
    float v = acc[i] + bb;
    float ge = 0.5f * v * (1.f + erff(v * 0.70710678118654752f));
    h1[(size_t)(m0 + m) * H_ + nt * 16 + n] = ge;
  }
}

// MLP head part 2: out[b] = h1[b,:] @ w2 + b2.
__global__ void __launch_bounds__(128)
out_kernel(const float* __restrict__ h1, const float* __restrict__ w2,
           const float* __restrict__ b2, float* __restrict__ out) {
  int b = threadIdx.x;
  float s = 0.f;
#pragma unroll 8
  for (int j = 0; j < H_; ++j) s += h1[(size_t)b * H_ + j] * w2[j];
  out[b] = s + b2[0];
}

extern "C" void kernel_launch(void* const* d_in, const int* in_sizes, int n_in,
                              void* d_out, int out_size, void* d_ws, size_t ws_size,
                              hipStream_t stream) {
  const float* x   = (const float*)d_in[0];
  const int*   mask= (const int*)d_in[1];
  const float* gk  = (const float*)d_in[2];   // (D, 3H)
  const float* grk = (const float*)d_in[3];   // (H, 3H)
  const float* gb  = (const float*)d_in[4];   // (2, 3H)
  const float* w1  = (const float*)d_in[5];   // (H, H)
  const float* b1  = (const float*)d_in[6];   // (H,)
  const float* w2  = (const float*)d_in[7];   // (H, 1)
  const float* b2  = (const float*)d_in[8];   // (1,)
  float* out = (float*)d_out;

  char* ws = (char*)d_ws;
  size_t off = 0;
  _Float16* xgp   = (_Float16*)(ws + off); off += (size_t)T_ * B_ * G_ * sizeof(_Float16);
  _Float16* xpack = (_Float16*)(ws + off); off += (size_t)T_ * B_ * D_ * sizeof(_Float16);
  _Float16* hfA   = (_Float16*)(ws + off); off += (size_t)B_ * H_ * sizeof(_Float16);
  _Float16* hfB   = (_Float16*)(ws + off); off += (size_t)B_ * H_ * sizeof(_Float16);
  _Float16* recKp = (_Float16*)(ws + off); off += (size_t)H_ * G_ * sizeof(_Float16);
  _Float16* gruKp = (_Float16*)(ws + off); off += (size_t)D_ * G_ * sizeof(_Float16);
  float*    hfin  = (float*)(ws + off);    off += (size_t)B_ * H_ * sizeof(float);
  float*    h1    = (float*)(ws + off);    off += (size_t)B_ * H_ * sizeof(float);
  unsigned int* bar = (unsigned int*)(ws + off); off += 256;
  (void)ws_size; (void)in_sizes; (void)n_in; (void)out_size;

  // One-time packing into WMMA fragment layouts (f16).
  pack_b_kernel<<<(H_ * G_) / 256, 256, 0, stream>>>(grk, recKp, H_, G_);
  pack_b_kernel<<<(D_ * G_) / 256, 256, 0, stream>>>(gk, gruKp, D_, G_);
  pack_a_x_kernel<<<(T_ * B_ * D_) / 256, 256, 0, stream>>>(x, xpack);
  // Zero h0 (f16 packed view) and the scan barrier counter (every call).
  zero_u32<<<(B_ * H_ * 2 / 4) / 256, 256, 0, stream>>>((unsigned int*)hfA, B_ * H_ * 2 / 4);
  zero_u32<<<1, 32, 0, stream>>>(bar, 1);

  // Phase 1: all input-gate projections in one big WMMA GEMM (2x2 tiles/wave).
  xg_gemm_kernel<<<dim3(T_ * (B_ / 16) / 2, G_ / 128), 128, 0, stream>>>(
      xpack, gruKp, gb, xgp);

  // Phase 2: the entire 512-step scan in one persistent kernel.
  gru_scan_kernel<<<NWG_SCAN, 32, 0, stream>>>(hfA, hfB, recKp, gb + G_, xgp,
                                               mask, hfin, bar);

  // Phase 3: MLP head.
  mlp1_kernel<<<dim3(B_ / 16, H_ / 64), 128, 0, stream>>>(hfin, w1, b1, h1);
  out_kernel<<<1, 128, 0, stream>>>(h1, w2, b2, out);
}
